// MultiHeadAttention_37005438223093
// MI455X (gfx1250) — matrix operand order repaired and checked by reading, not compiled, not run on hardware
//
#include <hip/hip_runtime.h>
#include <math.h>

// ---------------------------------------------------------------------------
// MHA forward for MI455X (gfx1250): bf16 WMMA (16x16x32) everywhere,
// f32 accumulation, flash-style fused attention, TDM-staged Q/V tiles,
// double-buffered GEMM.
// ---------------------------------------------------------------------------

typedef __bf16 bf16_t;
typedef __attribute__((ext_vector_type(16))) __bf16 v16bf;
typedef __attribute__((ext_vector_type(8)))  float  v8f;
typedef __attribute__((ext_vector_type(4)))  unsigned int u32x4;
typedef __attribute__((ext_vector_type(4)))  int i32x4;
typedef __attribute__((ext_vector_type(8)))  int i32x8;

#define BDIM 4
#define TDIM 2048
#define CDIM 1024
#define HDIM 16
#define DDIM 64
#define MDIM (BDIM * TDIM)   // 8192

#if defined(__has_builtin)
#  if __has_builtin(__builtin_amdgcn_tensor_load_to_lds) && \
      __has_builtin(__builtin_amdgcn_s_wait_tensorcnt)
#    define HAVE_TDM 1
#  endif
#endif
#ifndef HAVE_TDM
#  define HAVE_TDM 0
#endif

union FragB16 { v16bf v; u32x4 q[2]; };

__device__ __forceinline__ bf16_t cvt_bf16(float x)  { return (bf16_t)x; }
__device__ __forceinline__ bf16_t cvt_bf16(bf16_t x) { return x; }

#if HAVE_TDM
// ---------------------------------------------------------------------------
// Issue a TDM 2D tile load (bf16 elements) global -> LDS, per ISA D# layout
// (groups 0/1; groups 2/3 zero for 2D). Dims/strides in elements; lds pad
// codes per ISA: interval 4 -> 32 DWORDs, amount 3 -> 4 DWORDs.
// ---------------------------------------------------------------------------
__device__ __forceinline__ void tdm_load_2d_bf16(
    unsigned lds_addr, const void* gptr,
    unsigned tile_d0, unsigned tile_d1,
    unsigned tensor_d0, unsigned tensor_d1,
    unsigned long long stride0_elems,
    unsigned pad_enable, unsigned pad_interval_code, unsigned pad_amount_code)
{
    unsigned long long ga = (unsigned long long)gptr;
    u32x4 g0;
    g0[0] = 1u;                                               // count=1, user mode
    g0[1] = lds_addr;                                         // LDS byte address
    g0[2] = (unsigned)(ga & 0xFFFFFFFFu);                     // global_addr[31:0]
    g0[3] = (unsigned)((ga >> 32) & 0x1FFFFFFu) | (2u << 30); // ga[56:32] | type=2

    unsigned long long s0 = stride0_elems;
    unsigned long long s1 = stride0_elems * (unsigned long long)tensor_d1;
    i32x8 g1;
    g1[0] = (int)((1u << 16) |                 // data_size = 1 -> 2 bytes
                  (pad_enable << 20) |
                  (pad_interval_code << 22) |
                  (pad_amount_code << 25));
    g1[1] = (int)((tensor_d0 & 0xFFFFu) << 16);               // dim0[15:0]
    g1[2] = (int)(((tensor_d0 >> 16) & 0xFFFFu) |
                  ((tensor_d1 & 0xFFFFu) << 16));
    g1[3] = (int)(((tensor_d1 >> 16) & 0xFFFFu) |
                  ((tile_d0 & 0xFFFFu) << 16));
    g1[4] = (int)(tile_d1 & 0xFFFFu);                         // tile_dim1, tile_dim2=0
    g1[5] = (int)(s0 & 0xFFFFFFFFu);
    g1[6] = (int)(((s0 >> 32) & 0xFFFFu) | ((s1 & 0xFFFFu) << 16));
    g1[7] = (int)((s1 >> 16) & 0xFFFFFFFFu);

    i32x4 gz = {0, 0, 0, 0};
#if defined(__clang_major__) && (__clang_major__ >= 23)
    i32x8 gz8 = {0, 0, 0, 0, 0, 0, 0, 0};
    __builtin_amdgcn_tensor_load_to_lds(g0, g1, gz, gz, gz8, 0);
#else
    __builtin_amdgcn_tensor_load_to_lds(g0, g1, gz, gz, 0);
#endif
}
#endif  // HAVE_TDM

// ---------------------------------------------------------------------------
// GEMM: Out[M,N] = A[M,K] @ W[K,N] + bias.  Double-buffered LDS staging.
// OMODE 0: Out is bf16 written in split-head [B,H,T,D] layout.
// OMODE 1: Out is f32 row-major [M,N].
// Block: 256 threads (8 waves), 128x128 tile, K-step 32.
// Wave grid 2(M) x 4(N); each wave: 64x32 = 4x2 WMMA tiles.
// ---------------------------------------------------------------------------
template <typename AT, typename BT, int OMODE>
__global__ void __launch_bounds__(256)
gemm_wmma_kernel(const AT* __restrict__ A, const BT* __restrict__ W,
                 const float* __restrict__ bias, void* __restrict__ OutP,
                 int M, int N, int K)
{
    __shared__ __align__(16) bf16_t As[2][128][40];  // 128x32 + pad
    __shared__ __align__(16) bf16_t Bs[2][32][136];  // 32x128 + pad

    const int tid  = threadIdx.x;
    const int lane = tid & 31;
    const int wave = tid >> 5;
    const int wm   = wave & 1;
    const int wn   = wave >> 1;
    const int half = lane >> 4;
    const int l16  = lane & 15;

    const int m0 = blockIdx.y * 128;
    const int n0 = blockIdx.x * 128;

    v8f acc[4][2];
    #pragma unroll
    for (int i = 0; i < 4; ++i)
        #pragma unroll
        for (int j = 0; j < 2; ++j) acc[i][j] = {};

    auto stage = [&](int bufi, int k0) {
        int row = tid >> 1;
        int col = (tid & 1) * 16;
        const AT* srcA = A + (size_t)(m0 + row) * K + k0 + col;
        #pragma unroll
        for (int e = 0; e < 16; ++e) As[bufi][row][col + e] = cvt_bf16(srcA[e]);
        int rowb = tid >> 3;
        int colb = (tid & 7) * 16;
        const BT* srcB = W + (size_t)(k0 + rowb) * N + n0 + colb;
        #pragma unroll
        for (int e = 0; e < 16; ++e) Bs[bufi][rowb][colb + e] = cvt_bf16(srcB[e]);
    };

    stage(0, 0);
    int buf = 0;

    for (int k0 = 0; k0 < K; k0 += 32) {
        __syncthreads();   // staged tile visible; previous tile fully consumed

        // Fragment loads for this step (from As[buf]/Bs[buf])
        FragB16 af[4];
        #pragma unroll
        for (int mt = 0; mt < 4; ++mt) {
            int r = wm * 64 + mt * 16 + l16;
            af[mt].q[0] = *(const u32x4*)&As[buf][r][8 * half];
            af[mt].q[1] = *(const u32x4*)&As[buf][r][16 + 8 * half];
        }
        FragB16 bfr[2];
        #pragma unroll
        for (int nt = 0; nt < 2; ++nt) {
            int c  = wn * 32 + nt * 16 + l16;
            #pragma unroll
            for (int s = 0; s < 16; ++s)
                bfr[nt].v[s] = Bs[buf][(s & 7) + 8 * half + 16 * (s >> 3)][c];
        }

        // Prefetch next tile into the alternate buffer (overlaps with WMMAs)
        if (k0 + 32 < K) stage(buf ^ 1, k0 + 32);

        #pragma unroll
        for (int mt = 0; mt < 4; ++mt)
            #pragma unroll
            for (int nt = 0; nt < 2; ++nt)
                acc[mt][nt] = __builtin_amdgcn_wmma_f32_16x16x32_bf16(
                    false, af[mt].v, false, bfr[nt].v, (short)0, acc[mt][nt],
                    false, false);
        buf ^= 1;
    }

    // Epilogue. C layout: VGPR i -> row i + 8*half, col = lane&15.
    #pragma unroll
    for (int mt = 0; mt < 4; ++mt) {
        #pragma unroll
        for (int nt = 0; nt < 2; ++nt) {
            #pragma unroll
            for (int i = 0; i < 8; ++i) {
                int r = m0 + wm * 64 + mt * 16 + i + 8 * half;
                int c = n0 + wn * 32 + nt * 16 + l16;
                float v = acc[mt][nt][i] + bias[c];
                if constexpr (OMODE == 0) {
                    bf16_t* Out = (bf16_t*)OutP;
                    int b = r >> 11, t = r & (TDIM - 1);
                    int h = c >> 6,  d = c & (DDIM - 1);
                    Out[(((size_t)b * HDIM + h) * TDIM + t) * DDIM + d] = (bf16_t)v;
                } else {
                    float* Out = (float*)OutP;
                    Out[(size_t)r * N + c] = v;
                }
            }
        }
    }
}

// ---------------------------------------------------------------------------
// Fused attention: softmax(Q K^T / sqrt(D)) V, flash-style online softmax.
// Grid: (T/64, B*H). Block: 128 threads = 4 waves; wave owns 16 query rows.
// KV tile = 128 keys. Q and V tiles staged by the Tensor Data Mover (TDM);
// K is staged transposed by the VALU path (TDM cannot transpose), which
// overlaps with the V DMA.
// ---------------------------------------------------------------------------
__global__ void __launch_bounds__(128)
attn_wmma_kernel(const bf16_t* __restrict__ Q, const bf16_t* __restrict__ Kx,
                 const bf16_t* __restrict__ V, bf16_t* __restrict__ Ctx)
{
    __shared__ __align__(16) bf16_t Qs[64][72];      // [query][d], pad 8
    __shared__ __align__(16) bf16_t Ks[64][136];     // TRANSPOSED: [d][key]
    __shared__ __align__(16) bf16_t Vs[128][72];     // [key][d], pad 8
    __shared__ __align__(16) bf16_t Ps[4][16][136];  // per-wave P tile

    const int tid  = threadIdx.x;
    const int lane = tid & 31;
    const int wave = tid >> 5;
    const int half = lane >> 4;
    const int l16  = lane & 15;

    const int bh = blockIdx.y;
    const int q0 = blockIdx.x * 64;
    const int b  = bh >> 4;
    const int h  = bh & (HDIM - 1);

    const bf16_t* Qb = Q  + (size_t)bh * TDIM * DDIM;
    const bf16_t* Kb = Kx + (size_t)bh * TDIM * DDIM;
    const bf16_t* Vb = V  + (size_t)bh * TDIM * DDIM;

#if HAVE_TDM
    // Q tile 64x64 via TDM into padded Qs (row = 32 DW, pad = 4 DW).
    if (wave == 0) {
        tdm_load_2d_bf16((unsigned)(size_t)&Qs[0][0], Qb + (size_t)q0 * DDIM,
                         /*tile*/ DDIM, 64, /*tensor*/ DDIM, TDIM,
                         /*stride0*/ DDIM, /*pad*/ 1, 4, 3);
    }
#else
    {
        int row = tid >> 1;
        int col = (tid & 1) * 32;
        const bf16_t* src = Qb + (size_t)(q0 + row) * DDIM + col;
        #pragma unroll
        for (int e = 0; e < 32; ++e) Qs[row][col + e] = src[e];
    }
#endif

    const float scale = 0.125f;      // 1/sqrt(64)
    float mrow[8], lrow[8];
    #pragma unroll
    for (int i = 0; i < 8; ++i) { mrow[i] = -1e30f; lrow[i] = 0.f; }
    v8f o[4];
    #pragma unroll
    for (int i = 0; i < 4; ++i) o[i] = {};

    for (int kt = 0; kt < TDIM; kt += 128) {
        __syncthreads();   // previous K/V tiles fully consumed
#if HAVE_TDM
        if (wave == 0) {   // V tile 128x64 via TDM (one DMA per workgroup)
            tdm_load_2d_bf16((unsigned)(size_t)&Vs[0][0], Vb + (size_t)kt * DDIM,
                             /*tile*/ DDIM, 128, /*tensor*/ DDIM, TDIM,
                             /*stride0*/ DDIM, /*pad*/ 1, 4, 3);
        }
#endif
        {   // stage K transposed: one key row per thread (overlaps the V DMA)
            const bf16_t* src = Kb + (size_t)(kt + tid) * DDIM;
            #pragma unroll
            for (int d = 0; d < 64; ++d) Ks[d][tid] = src[d];
        }
#if !HAVE_TDM
        {   // fallback: stage V through VGPRs
            const bf16_t* src = Vb + (size_t)(kt + tid) * DDIM;
            #pragma unroll
            for (int d = 0; d < 64; ++d) Vs[tid][d] = src[d];
        }
#endif
#if HAVE_TDM
        if (wave == 0) __builtin_amdgcn_s_wait_tensorcnt(0);
#endif
        __syncthreads();

        // ---- S = Q K^T (16 rows x 128 keys per wave), contract over D=64 ----
        v8f s[8];
        #pragma unroll
        for (int nt = 0; nt < 8; ++nt) s[nt] = {};
        #pragma unroll
        for (int kk = 0; kk < 64; kk += 32) {
            FragB16 aq;
            int r = wave * 16 + l16;
            aq.q[0] = *(const u32x4*)&Qs[r][kk + 8 * half];
            aq.q[1] = *(const u32x4*)&Qs[r][kk + 16 + 8 * half];
            #pragma unroll
            for (int nt = 0; nt < 8; ++nt) {
                FragB16 bk;
                #pragma unroll
                for (int s = 0; s < 16; ++s)
                    bk.v[s] = Ks[kk + (s & 7) + 8 * half + 16 * (s >> 3)][nt * 16 + l16];
                s[nt] = __builtin_amdgcn_wmma_f32_16x16x32_bf16(
                    false, aq.v, false, bk.v, (short)0, s[nt], false, false);
            }
        }

        // ---- online softmax; row (i + 8*half) lives in lanes of this half ----
        #pragma unroll
        for (int i = 0; i < 8; ++i) {
            float rmax = -1e30f;
            #pragma unroll
            for (int nt = 0; nt < 8; ++nt) rmax = fmaxf(rmax, s[nt][i]);
            #pragma unroll
            for (int off = 1; off < 16; off <<= 1)
                rmax = fmaxf(rmax, __shfl_xor(rmax, off, 32));
            rmax *= scale;
            float mnew  = fmaxf(mrow[i], rmax);
            float alpha = __expf(mrow[i] - mnew);
            mrow[i] = mnew;
            float rsum = 0.f;
            #pragma unroll
            for (int nt = 0; nt < 8; ++nt) {
                float p = __expf(s[nt][i] * scale - mnew);
                rsum += p;
                Ps[wave][i + 8 * half][nt * 16 + l16] = (bf16_t)p;
            }
            #pragma unroll
            for (int off = 1; off < 16; off <<= 1)
                rsum += __shfl_xor(rsum, off, 32);
            lrow[i] = lrow[i] * alpha + rsum;
            #pragma unroll
            for (int nt = 0; nt < 4; ++nt) o[nt][i] *= alpha;
        }
        // Ps is private to this wave; LDS ops are in-order within a wave.

        // ---- O += P V, contract over 128 keys ----
        #pragma unroll
        for (int kk = 0; kk < 128; kk += 32) {
            FragB16 ap;
            ap.q[0] = *(const u32x4*)&Ps[wave][l16][kk + 8 * half];
            ap.q[1] = *(const u32x4*)&Ps[wave][l16][kk + 16 + 8 * half];
            #pragma unroll
            for (int nt = 0; nt < 4; ++nt) {
                FragB16 bv;
                #pragma unroll
                for (int s = 0; s < 16; ++s)
                    bv.v[s] = Vs[kk + (s & 7) + 8 * half + 16 * (s >> 3)][nt * 16 + l16];
                o[nt] = __builtin_amdgcn_wmma_f32_16x16x32_bf16(
                    false, ap.v, false, bv.v, (short)0, o[nt], false, false);
            }
        }
    }

    // ---- write context in [B,T,C] (merge heads) as bf16 ----
    #pragma unroll
    for (int nt = 0; nt < 4; ++nt) {
        #pragma unroll
        for (int i = 0; i < 8; ++i) {
            int t = q0 + wave * 16 + i + 8 * half;
            int d = nt * 16 + l16;
            float v = o[nt][i] / lrow[i];
            Ctx[((size_t)b * TDIM + t) * CDIM + h * DDIM + d] = (bf16_t)v;
        }
    }
}

// ---------------------------------------------------------------------------
extern "C" void kernel_launch(void* const* d_in, const int* in_sizes, int n_in,
                              void* d_out, int out_size, void* d_ws, size_t ws_size,
                              hipStream_t stream)
{
    const float* x  = (const float*)d_in[0];
    const float* Wq = (const float*)d_in[1];
    const float* bq = (const float*)d_in[2];
    const float* Wk = (const float*)d_in[3];
    const float* bk = (const float*)d_in[4];
    const float* Wv = (const float*)d_in[5];
    const float* bv = (const float*)d_in[6];
    const float* Wp = (const float*)d_in[7];
    const float* bp = (const float*)d_in[8];

    const size_t nqkv = (size_t)BDIM * HDIM * TDIM * DDIM;  // 8,388,608
    bf16_t* qbuf = (bf16_t*)d_ws;
    bf16_t* kbuf = qbuf + nqkv;
    bf16_t* vbuf = kbuf + nqkv;
    bf16_t* ctx  = vbuf + nqkv;     // total 64 MB of workspace

    dim3 ggrid(CDIM / 128, MDIM / 128);   // (8, 64)
    dim3 gblk(256);
    gemm_wmma_kernel<float, float, 0><<<ggrid, gblk, 0, stream>>>(x, Wq, bq, qbuf, MDIM, CDIM, CDIM);
    gemm_wmma_kernel<float, float, 0><<<ggrid, gblk, 0, stream>>>(x, Wk, bk, kbuf, MDIM, CDIM, CDIM);
    gemm_wmma_kernel<float, float, 0><<<ggrid, gblk, 0, stream>>>(x, Wv, bv, vbuf, MDIM, CDIM, CDIM);

    dim3 agrid(TDIM / 64, BDIM * HDIM);   // (32, 64)
    attn_wmma_kernel<<<agrid, dim3(128), 0, stream>>>(qbuf, kbuf, vbuf, ctx);

    gemm_wmma_kernel<bf16_t, float, 1><<<ggrid, gblk, 0, stream>>>(ctx, Wp, bp, d_out, MDIM, CDIM, CDIM);
}
